// DiagonalSSM_85478439125745
// MI455X (gfx1250) — matrix operand order repaired and checked by reading, not compiled, not run on hardware
//
#include <hip/hip_runtime.h>
#include <hip/hip_bf16.h>

// ---------- types ----------
typedef __bf16 bf16_t;
typedef __attribute__((ext_vector_type(16))) __bf16 v16bf;
typedef __attribute__((ext_vector_type(8)))  __bf16 v8bf;
typedef __attribute__((ext_vector_type(4)))  __bf16 v4bf;
typedef __attribute__((ext_vector_type(8)))  float  v8f;

// Problem dims (fixed by the reference)
#define BSZ   8
#define SLEN  2048
#define DIN   1024
#define DST   1024
#define DOUT  1024
#define MROWS (BSZ * SLEN)   // 16384

// ---------- CDNA5 hardware tanh (TRANS op) ----------
__device__ __forceinline__ float fast_tanhf(float x) {
#if __has_builtin(__builtin_amdgcn_tanhf)
  return __builtin_amdgcn_tanhf(x);
#else
  float r;
  asm("v_tanh_f32 %0, %1" : "=v"(r) : "v"(x));
  return r;
#endif
}

// ---------- fp32 -> bf16 conversion (4-wide) ----------
__global__ void cvt_f32_bf16_x4(const float* __restrict__ src,
                                bf16_t* __restrict__ dst, int n4) {
  int i = blockIdx.x * blockDim.x + threadIdx.x;
  if (i < n4) {
    float4 v = reinterpret_cast<const float4*>(src)[i];
    v4bf o;
    o.x = (bf16_t)v.x; o.y = (bf16_t)v.y; o.z = (bf16_t)v.z; o.w = (bf16_t)v.w;
    reinterpret_cast<v4bf*>(dst)[i] = o;
  }
}

// ---------- WMMA GEMM: C[M,N] = A[M,K] * B[N,K]^T (+bias) ----------
// All dims are template constants so fragment loads and the C stores fold
// into one base pointer + 24-bit immediate offsets (no per-store v_mul_u64).
// Workgroup: 256 threads = 8 waves arranged 4(M) x 2(N); each wave computes a
// 32x64 tile as 2x4 WMMA accumulators, K advanced by 32 per step.
// Intra-WG fragment reuse (4 waves share B, 2 waves share A) hits WGP$,
// giving ~64 FLOP per L2 byte.
template <int M, int N, int K, bool HAS_BIAS>
__global__ __launch_bounds__(256)
void gemm_bf16_wmma(const bf16_t* __restrict__ A,
                    const bf16_t* __restrict__ B,
                    float* __restrict__ C,
                    const float* __restrict__ bias) {
  const int lane = threadIdx.x & 31;
  const int wave = threadIdx.x >> 5;
  const int wm   = wave & 3;          // 0..3
  const int wn   = wave >> 2;         // 0..1

  const int Mb = blockIdx.x * 128 + wm * 32;
  const int Nb = blockIdx.y * 128 + wn * 64;

  const int lr = lane & 15;           // row (A) / col (B) within 16-tile
  const int hk = lane >> 4;           // 0 or 1 (K half selector)

  v8f acc[2][4];
#pragma unroll
  for (int mt = 0; mt < 2; ++mt)
#pragma unroll
    for (int nt = 0; nt < 4; ++nt)
      acc[mt][nt] = (v8f){0.f, 0.f, 0.f, 0.f, 0.f, 0.f, 0.f, 0.f};

  // Per-lane base pointers; mt/nt deltas are compile-time immediates.
  const bf16_t* abase = A + (size_t)(Mb + lr) * K + hk * 8;
  const bf16_t* bbase = B + (size_t)(Nb + lr) * K + hk * 8;

  for (int k = 0; k < K; k += 32) {
    v16bf afrag[2];
#pragma unroll
    for (int mt = 0; mt < 2; ++mt) {
      const bf16_t* p = abase + (size_t)mt * 16 * K + k;
      v8bf lo = *reinterpret_cast<const v8bf*>(p);        // K = klo .. klo+7
      v8bf hi = *reinterpret_cast<const v8bf*>(p + 16);   // K = klo+16 .. +23
      afrag[mt] = __builtin_shufflevector(lo, hi,
          0, 1, 2, 3, 4, 5, 6, 7, 8, 9, 10, 11, 12, 13, 14, 15);
    }
    v16bf bfrag[4];
#pragma unroll
    for (int nt = 0; nt < 4; ++nt) {
      const bf16_t* p = bbase + (size_t)nt * 16 * K + k;
      v8bf lo = *reinterpret_cast<const v8bf*>(p);        // K = klo .. klo+7
      v8bf hi = *reinterpret_cast<const v8bf*>(p + 16);   // K = klo+16 .. +23
      bfrag[nt] = __builtin_shufflevector(lo, hi,
          0, 1, 2, 3, 4, 5, 6, 7, 8, 9, 10, 11, 12, 13, 14, 15);
    }
#pragma unroll
    for (int mt = 0; mt < 2; ++mt)
#pragma unroll
      for (int nt = 0; nt < 4; ++nt)
        acc[mt][nt] = __builtin_amdgcn_wmma_f32_16x16x32_bf16(
            /*neg_a=*/false, afrag[mt], /*neg_b=*/false, bfrag[nt],
            /*c_mod=*/(short)0, acc[mt][nt],
            /*reuse_a=*/false, /*reuse_b=*/false);
  }

  // C layout: VGPR r -> row (hk*8 + r), col = lane%16. One base pointer,
  // all 64 stores at compile-time immediate offsets.
  float* cbase = C + (size_t)(Mb + hk * 8) * N + Nb + lr;
#pragma unroll
  for (int nt = 0; nt < 4; ++nt) {
    float bv;
    if constexpr (HAS_BIAS) bv = bias[Nb + nt * 16 + lr];
#pragma unroll
    for (int mt = 0; mt < 2; ++mt) {
#pragma unroll
      for (int r = 0; r < 8; ++r) {
        float v = acc[mt][nt][r];
        if constexpr (HAS_BIAS) v += bv;
        cbase[(size_t)(mt * 16 + r) * N + nt * 16] = v;
      }
    }
  }
}

// ---------- sequential scan: state = tanh(decay*state + Bx[s]) ----------
// Only 8192 independent chains (~1 wave/SIMD), so latency must be hidden
// inside each thread: register double-buffered tiles of TSTEPS timesteps.
// All TSTEPS loads for tile t+1 are independent and issued (clause-batched)
// before the tanh chain of tile t runs (~TSTEPS * ~12cy covers HBM latency).
#define TSTEPS 64

__device__ __forceinline__ void scan_load_tile(float (&buf)[TSTEPS],
                                               const float* __restrict__ Bx,
                                               size_t base) {
#pragma unroll
  for (int j = 0; j < TSTEPS; ++j)
    buf[j] = Bx[base + (size_t)j * DST];
}

__device__ __forceinline__ void scan_compute_tile(const float (&buf)[TSTEPS],
                                                  bf16_t* __restrict__ H,
                                                  size_t base, float decay,
                                                  float& state) {
#pragma unroll
  for (int j = 0; j < TSTEPS; ++j) {
    state = fast_tanhf(__builtin_fmaf(decay, state, buf[j]));
    H[base + (size_t)j * DST] = (bf16_t)state;
  }
}

__global__ __launch_bounds__(256)
void scan_kernel(const float* __restrict__ A,
                 const float* __restrict__ Bx,   // (B,S,Ds) f32
                 bf16_t* __restrict__ H) {       // (B,S,Ds) bf16
  const int t = blockIdx.x * blockDim.x + threadIdx.x;
  const int b = t >> 10;            // / DST
  const int n = t & (DST - 1);      // % DST
  const float decay = __expf(-fminf(A[n], 10.f));

  const size_t base = (size_t)b * SLEN * DST + n;
  const size_t tstride = (size_t)TSTEPS * DST;
  constexpr int NT = SLEN / TSTEPS;   // 32 tiles (even)

  float buf0[TSTEPS], buf1[TSTEPS];
  float state = 0.f;

  scan_load_tile(buf0, Bx, base);                       // tile 0 in flight
  for (int tt = 0; tt < NT; tt += 2) {
    const size_t b0 = base + (size_t)tt * tstride;
    if (tt + 1 < NT) scan_load_tile(buf1, Bx, b0 + tstride);      // prefetch t+1
    scan_compute_tile(buf0, H, b0, decay, state);                 // compute t
    if (tt + 2 < NT) scan_load_tile(buf0, Bx, b0 + 2 * tstride);  // prefetch t+2
    if (tt + 1 < NT) scan_compute_tile(buf1, H, b0 + tstride, decay, state);
  }
}

// ---------- driver ----------
extern "C" void kernel_launch(void* const* d_in, const int* in_sizes, int n_in,
                              void* d_out, int out_size, void* d_ws, size_t ws_size,
                              hipStream_t stream) {
  const float* x  = (const float*)d_in[0];   // (8,2048,1024)
  const float* A  = (const float*)d_in[1];   // (1024,)
  const float* Bw = (const float*)d_in[2];   // (1024,1024)
  const float* Wo = (const float*)d_in[3];   // (1024,1024)
  const float* bo = (const float*)d_in[4];   // (1024,)

  // Workspace layout (bytes):
  //   xb  : bf16 x            32 MB
  //   Bwb : bf16 B_w           2 MB
  //   Wob : bf16 Wo            2 MB
  //   Bx  : f32 projection    64 MB
  //   Hb  : bf16 hidden       32 MB
  char* ws = (char*)d_ws;
  bf16_t* xb  = (bf16_t*)ws;
  bf16_t* Bwb = (bf16_t*)(ws + (size_t)MROWS * DIN * 2);
  bf16_t* Wob = Bwb + (size_t)DST * DIN;
  float*  Bx  = (float*)((char*)Wob + (size_t)DOUT * DST * 2);
  bf16_t* Hb  = (bf16_t*)((char*)Bx + (size_t)MROWS * DST * 4);

  // 1) convert inputs to bf16
  {
    int n4 = MROWS * DIN / 4;
    cvt_f32_bf16_x4<<<(n4 + 255) / 256, 256, 0, stream>>>(x, xb, n4);
    n4 = DST * DIN / 4;
    cvt_f32_bf16_x4<<<(n4 + 255) / 256, 256, 0, stream>>>(Bw, Bwb, n4);
    n4 = DOUT * DST / 4;
    cvt_f32_bf16_x4<<<(n4 + 255) / 256, 256, 0, stream>>>(Wo, Wob, n4);
  }

  // 2) Bx = x @ B_w^T  (M=16384, N=1024, K=1024), f32 accumulate
  {
    dim3 grid(MROWS / 128, DST / 128);
    gemm_bf16_wmma<MROWS, DST, DIN, false><<<grid, 256, 0, stream>>>(
        xb, Bwb, Bx, nullptr);
  }

  // 3) sequential scan -> Hb (bf16)
  scan_kernel<<<(BSZ * DST) / 256, 256, 0, stream>>>(A, Bx, Hb);

  // 4) out = Hb @ Wo^T + bo  -> d_out (f32)
  {
    dim3 grid(MROWS / 128, DOUT / 128);
    gemm_bf16_wmma<MROWS, DOUT, DST, true><<<grid, 256, 0, stream>>>(
        Hb, Wob, (float*)d_out, bo);
  }
}
